// PtrNet_tanh_14035953123644
// MI455X (gfx1250) — matrix operand order repaired and checked by reading, not compiled, not run on hardware
//
#include <hip/hip_runtime.h>

// PtrNet (B=1024, N=16, I=256, H=1024) on gfx1250.
// GEMMs via v_wmma_f32_16x16x32_bf16 (fp32 accumulate), weights converted
// to bf16 once per launch (L2-resident, ~19MB), activations converted
// fp32->bf16 through LDS. Each wave owns a 16x64 tile (4 accumulators) so
// 4 WMMA issue per barrier pair. Requires ~150MB of d_ws.

#define Bsz 1024
#define Nsz 16
#define Isz 256
#define Hsz 1024
#define H3  (3 * Hsz)

// padded LDS row stride for the A tile (32 K + 8 pad) -> conflict-free frag reads
#define APAD 40

typedef __attribute__((ext_vector_type(16))) __bf16 v16bf;
typedef __attribute__((ext_vector_type(8)))  float  v8f;
typedef __attribute__((ext_vector_type(4)))  int    v4i;

#define GAS __attribute__((address_space(1)))
#define LAS __attribute__((address_space(3)))

union ABfrag { v16bf v; uint4 q[2]; };

#if defined(__HIP_DEVICE_COMPILE__) && \
    __has_builtin(__builtin_amdgcn_global_load_async_to_lds_b128) && \
    __has_builtin(__builtin_amdgcn_s_wait_asynccnt)
#define USE_ASYNC_LDS 1
#else
#define USE_ASYNC_LDS 0
#endif

// ---------------------------------------------------------------------------
// Generic WMMA GEMM: C[M x Nout] = A[M x K](f32, row stride lda) * W[Nout x K]^T
//                    (+ bias[Nout]) (optional rowscale[m] applied to A rows)
// Output either f32 (C) or bf16 (Cb).
// Block = 256 thr = 8 waves (2 M x 4 N) -> 32 x 256 tile; wave = 16 x 64.
// ---------------------------------------------------------------------------
__global__ __launch_bounds__(256) void gemm_wmma_bf16(
    const float* __restrict__ A, int lda,
    const __bf16* __restrict__ W,
    const float* __restrict__ bias,
    const float* __restrict__ rowscale,
    float* __restrict__ C,
    __bf16* __restrict__ Cb,
    int M, int Nout, int K)
{
    __shared__ __bf16 As[32 * APAD];   // 32 rows x 32 K (bf16) + pad

    const int tid  = threadIdx.x;
    const int lane = tid & 31;
    const int wave = tid >> 5;            // 0..7
    const int mi   = wave >> 2;           // 0..1
    const int ni   = wave & 3;            // 0..3
    const int l15  = lane & 15;
    const int hi   = lane >> 4;           // 0 or 1

    const int m0 = blockIdx.y * 32;
    const int n0 = blockIdx.x * 256;

    // cooperative A staging: 32 rows x 32 cols, 4 f32 per thread
    const int sr = tid >> 3;              // 0..31
    const int sc = (tid & 7) << 2;        // 0,4,...,28
    const float* Arow  = A + (size_t)(m0 + sr) * (size_t)lda;
    const float  scale = rowscale ? rowscale[m0 + sr] : 1.0f;

    // this wave's 4 weight rows (= output columns) for the B fragments
    const int wcol0 = n0 + ni * 64 + l15;
    const __bf16* Wp[4];
    #pragma unroll
    for (int g = 0; g < 4; ++g)
        Wp[g] = W + (size_t)(wcol0 + 16 * g) * (size_t)K;

    v8f acc[4] = {};
    for (int k0 = 0; k0 < K; k0 += 32) {
        // ---- stage A tile (f32 -> bf16) into LDS ----
        float4 av = *(const float4*)(Arow + k0 + sc);
        union { __bf16 h[4]; ushort4 u; } pk;
        pk.h[0] = (__bf16)(av.x * scale);
        pk.h[1] = (__bf16)(av.y * scale);
        pk.h[2] = (__bf16)(av.z * scale);
        pk.h[3] = (__bf16)(av.w * scale);
        *(ushort4*)&As[sr * APAD + sc] = pk.u;
        __syncthreads();

        // ---- A fragment (16x32, ISA 16-bit A layout), reused 4x ----
        ABfrag a;
        const int ar = (mi * 16 + l15) * APAD;
        a.q[0] = *(const uint4*)&As[ar + hi * 8];        // K = kb..kb+7
        a.q[1] = *(const uint4*)&As[ar + 16 + hi * 8];   // K = 16+kb..16+kb+7

        __builtin_prefetch(Wp[0] + k0 + 128, 0, 1);      // global_prefetch_b8

        // ---- 4 B fragments (32x16 each): K = kb..kb+7 and 16+kb..16+kb+7 per lane ----
        #pragma unroll
        for (int g = 0; g < 4; ++g) {
            ABfrag b;
            const __bf16* wp = Wp[g] + k0 + hi * 8;
            b.q[0] = *(const uint4*)(wp);
            b.q[1] = *(const uint4*)(wp + 16);
            acc[g] = __builtin_amdgcn_wmma_f32_16x16x32_bf16(
                false, a.v, false, b.v, (short)0, acc[g], false, false);
        }
        __syncthreads();
    }

    // ---- epilogue: C/D layout -> lane holds col, 8 rows ----
    const int mrow = m0 + mi * 16 + hi * 8;
    #pragma unroll
    for (int g = 0; g < 4; ++g) {
        const int col = wcol0 + 16 * g;
        const float bv = bias ? bias[col] : 0.0f;
        if (Cb) {
            #pragma unroll
            for (int r = 0; r < 8; ++r)
                Cb[(size_t)(mrow + r) * (size_t)Nout + col] = (__bf16)(acc[g][r] + bv);
        } else {
            #pragma unroll
            for (int r = 0; r < 8; ++r)
                C[(size_t)(mrow + r) * (size_t)Nout + col] = acc[g][r] + bv;
        }
    }
}

// ---------------------------------------------------------------------------
// small helper kernels
// ---------------------------------------------------------------------------
__global__ void zero_f32(float* __restrict__ p, int n) {
    int i = blockIdx.x * 256 + threadIdx.x;
    if (i < n) p[i] = 0.0f;
}

__global__ void cvt_bf16(const float* __restrict__ s, __bf16* __restrict__ d, int n) {
    int i = blockIdx.x * 256 + threadIdx.x;
    if (i < n) d[i] = (__bf16)s[i];
}

// d[c*rows + r] = s[r*cols + c]  (store W^T so GEMM sees [Nout x K] row-major)
__global__ void cvt_bf16_T(const float* __restrict__ s, __bf16* __restrict__ d,
                           int rows, int cols) {
    int i = blockIdx.x * 256 + threadIdx.x;
    if (i < rows * cols) {
        int r = i / cols, c = i - r * cols;
        d[(size_t)c * rows + r] = (__bf16)s[i];
    }
}

// t_gate[n*B+b] and idx_seq[t*B+b]
__global__ void prep_tg_idx(const float* __restrict__ phis,
                            float* __restrict__ tg, int* __restrict__ idxseq) {
    int b = blockIdx.x * 256 + threadIdx.x;
    if (b >= Bsz) return;
    tg[b] = 1.0f;
    #pragma unroll
    for (int n = 1; n < Nsz; ++n)
        tg[n * Bsz + b] = phis[((size_t)b * Nsz + n) * Nsz + (n - 1)];
    for (int t = 0; t < Nsz; ++t) {
        const float* pr = phis + ((size_t)b * Nsz + t) * Nsz;
        float s = 0.f;
        #pragma unroll
        for (int k = 0; k < Nsz; ++k) s += pr[k];
        int si = (int)(s + 0.5f);                 // phis entries are 0/1, diag=1
        idxseq[t * Bsz + b] = (si + t - 1) & (Nsz - 1);
    }
}

// GRU gate fusion: h = (1-z)*g + z*h_in ; optionally h_in = rowscale[b]*h.
// Optionally mirror new h into henc[:, t, :].
__global__ void gru_gates(const float* __restrict__ gi, const float* __restrict__ gh,
                          float* __restrict__ h, const float* __restrict__ rowscale,
                          float* __restrict__ henc, int t) {
    int i = blockIdx.x * 256 + threadIdx.x;       // < B*H
    int b = i >> 10, j = i & (Hsz - 1);
    const float* gib = gi + (size_t)b * H3;
    const float* ghb = gh + (size_t)b * H3;
    float ir = gib[j], iz = gib[Hsz + j], ig = gib[2 * Hsz + j];
    float hr = ghb[j], hz = ghb[Hsz + j], hg = ghb[2 * Hsz + j];
    float hin = h[i];
    if (rowscale) hin *= rowscale[b];
    float r = 1.0f / (1.0f + expf(-(ir + hr)));
    float z = 1.0f / (1.0f + expf(-(iz + hz)));
    float g = tanhf(ig + r * hg);
    float hn = (1.0f - z) * g + z * hin;
    h[i] = hn;
    if (henc) henc[((size_t)b * Nsz + t) * Hsz + j] = hn;
}

// decoder init: hidden = henc[b, idx_seq[0][b]], inp = init_token
__global__ void dec_init(const float* __restrict__ henc, const int* __restrict__ idxseq,
                         const float* __restrict__ tok,
                         float* __restrict__ h, float* __restrict__ inp) {
    int i = blockIdx.x * 256 + threadIdx.x;       // < B*(H+I)
    int b = i / (Hsz + Isz), j = i - b * (Hsz + Isz);
    if (j < Hsz) {
        int idx = idxseq[b];
        h[(size_t)b * Hsz + j] = henc[((size_t)b * Nsz + idx) * Hsz + j];
    } else {
        int jj = j - Hsz;
        inp[(size_t)b * Isz + jj] = tok[jj];
    }
}

// decoder reset: x = t*x + (1-t)*init  (hidden from henc[idx], inp from token)
__global__ void dec_reset(const float* __restrict__ henc, const int* __restrict__ idxseq,
                          const float* __restrict__ tok, const float* __restrict__ tg,
                          float* __restrict__ h, float* __restrict__ inp, int t) {
    int i = blockIdx.x * 256 + threadIdx.x;       // < B*(H+I)
    int b = i / (Hsz + Isz), j = i - b * (Hsz + Isz);
    float tv = tg[t * Bsz + b];
    if (j < Hsz) {
        int idx = idxseq[t * Bsz + b];
        float ih = henc[((size_t)b * Nsz + idx) * Hsz + j];
        float* hp = &h[(size_t)b * Hsz + j];
        *hp = tv * (*hp) + (1.0f - tv) * ih;
    } else {
        int jj = j - Hsz;
        float* ip = &inp[(size_t)b * Isz + jj];
        *ip = tv * (*ip) + (1.0f - tv) * tok[jj];
    }
}

// attention: u[n] = sum_h tanh(W1xe[b,n,h] + hW2[b,h]) * v[h];
// masked softmax over n (mask = phis[b,t,:]); write attn to out[b,t,:];
// inp[b,i] = sum_n attn[n]*input[b,n,i].   one block (256 thr) per b.
// hW2[b,:] and v[:] are each reused 16x within the block -> stage in LDS
// (async global->LDS copy on gfx1250 when the builtin is available).
__global__ __launch_bounds__(256) void attn_step(
    const __bf16* __restrict__ W1xe, const float* __restrict__ hW2,
    const float* __restrict__ v, const float* __restrict__ phis,
    const float* __restrict__ input, float* __restrict__ out,
    float* __restrict__ inp, int t)
{
    __shared__ float sHW[Hsz];
    __shared__ float sV[Hsz];
    __shared__ float sRed[256];
    __shared__ float sA[Nsz];
    const int b = blockIdx.x, tid = threadIdx.x;
    const int n = tid >> 4, l16 = tid & 15;

    const float* hw = hW2 + (size_t)b * Hsz;
    const int tid4 = tid * 4;                     // 256 thr x 4 f32 = 1024
#if USE_ASYNC_LDS
    __builtin_amdgcn_global_load_async_to_lds_b128(
        (GAS v4i*)(hw + tid4), (LAS v4i*)&sHW[tid4], 0, 0);
    __builtin_amdgcn_global_load_async_to_lds_b128(
        (GAS v4i*)(v + tid4), (LAS v4i*)&sV[tid4], 0, 0);
    __builtin_amdgcn_s_wait_asynccnt(0);
#else
    *(float4*)&sHW[tid4] = *(const float4*)(hw + tid4);
    *(float4*)&sV[tid4]  = *(const float4*)(v + tid4);
#endif
    __syncthreads();

    const __bf16* w1 = W1xe + ((size_t)b * Nsz + n) * Hsz;
    float p = 0.f;
    for (int hh = l16; hh < Hsz; hh += 16)
        p += tanhf((float)w1[hh] + sHW[hh]) * sV[hh];
    sRed[tid] = p;
    __syncthreads();
    if (tid < Nsz) {
        float u = 0.f;
        #pragma unroll
        for (int i = 0; i < 16; ++i) u += sRed[tid * 16 + i];
        sRed[tid] = u;
    }
    __syncthreads();
    if (tid == 0) {
        float um[Nsz], msk[Nsz];
        float m = -3.0e38f;
        #pragma unroll
        for (int k = 0; k < Nsz; ++k) {
            msk[k] = phis[((size_t)b * Nsz + t) * Nsz + k];
            um[k]  = sRed[k] * msk[k];
            m = fmaxf(m, um[k]);
        }
        float s = 0.f;
        #pragma unroll
        for (int k = 0; k < Nsz; ++k) {
            float e = expf(um[k] - m) * msk[k];
            sA[k] = e; s += e;
        }
        float inv = 1.0f / s;
        #pragma unroll
        for (int k = 0; k < Nsz; ++k) {
            float a = sA[k] * inv;
            sA[k] = a;
            out[((size_t)b * Nsz + t) * Nsz + k] = a;
        }
    }
    __syncthreads();
    // context vector (I == 256 == blockDim)
    float acc = 0.f;
    #pragma unroll
    for (int k = 0; k < Nsz; ++k)
        acc += sA[k] * input[((size_t)b * Nsz + k) * Isz + tid];
    inp[(size_t)b * Isz + tid] = acc;
}

// ---------------------------------------------------------------------------
extern "C" void kernel_launch(void* const* d_in, const int* in_sizes, int n_in,
                              void* d_out, int out_size, void* d_ws, size_t ws_size,
                              hipStream_t stream) {
    const float* input = (const float*)d_in[0];   // (B,N,I)
    const float* phis  = (const float*)d_in[1];   // (B,N,N)
    const float* tok   = (const float*)d_in[2];   // (I)
    const float* W1    = (const float*)d_in[3];   // (H,H)
    const float* W2    = (const float*)d_in[4];   // (H,H)
    const float* vv    = (const float*)d_in[5];   // (H,1)
    const float* Wih_e = (const float*)d_in[6];   // (3H,I)
    const float* Whh_e = (const float*)d_in[7];   // (3H,H)
    const float* bih_e = (const float*)d_in[8];
    const float* bhh_e = (const float*)d_in[9];
    const float* Wih_d = (const float*)d_in[10];
    const float* Whh_d = (const float*)d_in[11];
    const float* bih_d = (const float*)d_in[12];
    const float* bhh_d = (const float*)d_in[13];
    float* out = (float*)d_out;                   // (B,N,N)

    // workspace carve-up (256B aligned), ~150 MB total
    char* base = (char*)d_ws;
    size_t off = 0;
    auto carve = [&](size_t bytes) -> void* {
        void* p = base + off;
        off = (off + bytes + 255) & ~(size_t)255;
        return p;
    };
    __bf16* wWih_e = (__bf16*)carve((size_t)H3 * Isz * 2);
    __bf16* wWhh_e = (__bf16*)carve((size_t)H3 * Hsz * 2);
    __bf16* wWih_d = (__bf16*)carve((size_t)H3 * Isz * 2);
    __bf16* wWhh_d = (__bf16*)carve((size_t)H3 * Hsz * 2);
    __bf16* wW1t   = (__bf16*)carve((size_t)Hsz * Hsz * 2);
    __bf16* wW2t   = (__bf16*)carve((size_t)Hsz * Hsz * 2);
    float*  tg     = (float*)carve((size_t)Nsz * Bsz * 4);
    int*    idxs   = (int*)  carve((size_t)Nsz * Bsz * 4);
    float*  h      = (float*)carve((size_t)Bsz * Hsz * 4);
    float*  henc   = (float*)carve((size_t)Bsz * Nsz * Hsz * 4);
    __bf16* W1xe   = (__bf16*)carve((size_t)Bsz * Nsz * Hsz * 2);
    float*  gi     = (float*)carve((size_t)Bsz * H3 * 4);
    float*  gh     = (float*)carve((size_t)Bsz * H3 * 4);
    float*  hW2    = (float*)carve((size_t)Bsz * Hsz * 4);
    float*  inp    = (float*)carve((size_t)Bsz * Isz * 4);
    (void)ws_size; (void)in_sizes; (void)n_in; (void)out_size;

    // ---- one-time prep (per launch; deterministic) ----
    cvt_bf16<<<(H3 * Isz) / 256, 256, 0, stream>>>(Wih_e, wWih_e, H3 * Isz);
    cvt_bf16<<<(H3 * Hsz) / 256, 256, 0, stream>>>(Whh_e, wWhh_e, H3 * Hsz);
    cvt_bf16<<<(H3 * Isz) / 256, 256, 0, stream>>>(Wih_d, wWih_d, H3 * Isz);
    cvt_bf16<<<(H3 * Hsz) / 256, 256, 0, stream>>>(Whh_d, wWhh_d, H3 * Hsz);
    cvt_bf16_T<<<(Hsz * Hsz) / 256, 256, 0, stream>>>(W1, wW1t, Hsz, Hsz);
    cvt_bf16_T<<<(Hsz * Hsz) / 256, 256, 0, stream>>>(W2, wW2t, Hsz, Hsz);
    prep_tg_idx<<<Bsz / 256, 256, 0, stream>>>(phis, tg, idxs);
    zero_f32<<<(Bsz * Hsz) / 256, 256, 0, stream>>>(h, Bsz * Hsz);

    const dim3 g3h(H3 / 256, Bsz / 32);    // 12 x 32
    const dim3 gHH(Hsz / 256, Bsz / 32);   //  4 x 32

    // ---- encoder scan ----
    for (int t = 0; t < Nsz; ++t) {
        gemm_wmma_bf16<<<g3h, 256, 0, stream>>>(input + (size_t)t * Isz, Nsz * Isz,
            wWih_e, bih_e, nullptr, gi, nullptr, Bsz, H3, Isz);
        gemm_wmma_bf16<<<g3h, 256, 0, stream>>>(h, Hsz,
            wWhh_e, bhh_e, tg + t * Bsz, gh, nullptr, Bsz, H3, Hsz);
        gru_gates<<<(Bsz * Hsz) / 256, 256, 0, stream>>>(gi, gh, h, tg + t * Bsz, henc, t);
    }

    // ---- W1xe = henc @ W1 (bf16 out), decoder init ----
    gemm_wmma_bf16<<<dim3(Hsz / 256, (Bsz * Nsz) / 32), 256, 0, stream>>>(
        henc, Hsz, wW1t, nullptr, nullptr, nullptr, W1xe, Bsz * Nsz, Hsz, Hsz);
    dec_init<<<(Bsz * (Hsz + Isz)) / 256, 256, 0, stream>>>(henc, idxs, tok, h, inp);

    // ---- decoder scan ----
    for (int t = 0; t < Nsz; ++t) {
        dec_reset<<<(Bsz * (Hsz + Isz)) / 256, 256, 0, stream>>>(henc, idxs, tok, tg, h, inp, t);
        gemm_wmma_bf16<<<g3h, 256, 0, stream>>>(inp, Isz,
            wWih_d, bih_d, nullptr, gi, nullptr, Bsz, H3, Isz);
        gemm_wmma_bf16<<<g3h, 256, 0, stream>>>(h, Hsz,
            wWhh_d, bhh_d, nullptr, gh, nullptr, Bsz, H3, Hsz);
        gru_gates<<<(Bsz * Hsz) / 256, 256, 0, stream>>>(gi, gh, h, nullptr, nullptr, t);
        gemm_wmma_bf16<<<gHH, 256, 0, stream>>>(h, Hsz,
            wW2t, nullptr, nullptr, hW2, nullptr, Bsz, Hsz, Hsz);
        attn_step<<<Bsz, 256, 0, stream>>>(W1xe, hW2, vv, phis, input, out, inp, t);
    }
}
